// MultiGraphEnsembleCNN_70806830842520
// MI455X (gfx1250) — matrix operand order repaired and checked by reading, not compiled, not run on hardware
//
#include <hip/hip_runtime.h>
#include <hip/hip_bf16.h>
#include <math.h>
#include <stdint.h>

typedef __attribute__((ext_vector_type(16))) _Float16 v16h;
typedef __attribute__((ext_vector_type(8)))  _Float16 v8h;
typedef __attribute__((ext_vector_type(8)))  float    v8f;

#define N_NODES 20000
#define D_IN    256
#define KG_EDGES 200000
#define SL_EDGES 20000
#define NE_OUT  40000   // 2*SL_EDGES scored edges

// ---------------------------------------------------------------- utilities

__global__ void fill_f32_kernel(float* __restrict__ p, float v, int n) {
  int i = blockIdx.x * blockDim.x + threadIdx.x;
  if (i < n) p[i] = v;
}

__global__ void cvt_f32_to_f16_kernel(const float* __restrict__ s,
                                      _Float16* __restrict__ d, int n) {
  int i = blockIdx.x * blockDim.x + threadIdx.x;
  if (i < n) d[i] = (_Float16)s[i];
}

// deg[dst] += 1 for each edge (deg buffer pre-filled with 1.0 == self loop)
__global__ void deg_count_kernel(const int* __restrict__ edges, int E,
                                 float* __restrict__ deg) {
  int e = blockIdx.x * blockDim.x + threadIdx.x;
  if (e < E) atomicAdd(&deg[edges[E + e]], 1.0f);
}

// -------------------------------------------------- B fragment pre-packing
// Convert f32 weights B[K x Ncols] into WMMA-fragment order (f16):
//   Bf[((tn*(K/32) + kblk)*32 + lane)*16 + j] = B[(kblk*32 + bKoff(lane) + j + (j&8))*Ncols + tn*16 + (lane&15)]
// so the GEMM can read each lane's 16-half B fragment as one contiguous 32B load.
__global__ void pack_b_frags_kernel(const float* __restrict__ B,
                                    _Float16* __restrict__ Bf,
                                    int K, int Ncols) {
  int idx = blockIdx.x * blockDim.x + threadIdx.x;
  int total = K * Ncols;          // same element count, new order
  if (idx >= total) return;
  int j    = idx & 15;
  int lane = (idx >> 4) & 31;
  int blk  = idx >> 9;            // tn*(K/32) + kblk
  int kblks = K >> 5;
  int kblk = blk % kblks;
  int tn   = blk / kblks;
  int bKoff = (lane < 16) ? 0 : 8;
  int col   = tn * 16 + (lane & 15);
  Bf[idx] = (_Float16)B[(kblk * 32 + bKoff + j + (j & 8)) * Ncols + col];
}

// ---------------------------------------------------------------- WMMA GEMM
// C[M x Ncols] = A[M x K](f16) * Bf(packed f16 fragments), fp32 accumulate.
// One wave (32 threads) per 16x16 output tile; layouts per CDNA5 ISA 7.12.2.
__global__ void gemm_f16_wmma_kernel(const _Float16* __restrict__ A,
                                     const _Float16* __restrict__ Bf,
                                     float* __restrict__ C,
                                     int K, int Ncols) {
  const int lane  = threadIdx.x;            // 0..31
  const int tm    = blockIdx.x;
  const int tn    = blockIdx.y;
  const int row   = tm * 16 + (lane & 15);  // A row held by this lane
  const int aKoff = (lane < 16) ? 0 : 8;    // lanes 16-31 hold K+8..15 / K+24..31
  const int kblks = K >> 5;

  const _Float16* __restrict__ Arow = A + (size_t)row * K + aKoff;
  const _Float16* __restrict__ Bp =
      Bf + ((size_t)tn * kblks * 32 + lane) * 16;

  union AFrag { v16h v; struct { v8h lo, hi; } p; };

  v8f acc = {};
  for (int kblk = 0; kblk < kblks; ++kblk) {
    AFrag a;
    a.p.lo = *(const v8h*)(Arow + kblk * 32);        // K+0..7  (or +8..15)
    a.p.hi = *(const v8h*)(Arow + kblk * 32 + 16);   // K+16..23 (or +24..31)
    v16h b = *(const v16h*)(Bp + (size_t)kblk * 512);
    acc = __builtin_amdgcn_wmma_f32_16x16x32_f16(
        /*neg_a=*/false, a.v, /*neg_b=*/false, b,
        /*c_mod=*/(short)0, acc, /*reuse_a=*/false, /*reuse_b=*/false);
  }
  // C/D layout: VGPR r -> (M = r or r+8, N = lane&15)
  const int mbase = tm * 16 + ((lane < 16) ? 0 : 8);
  const int col   = tn * 16 + (lane & 15);
#pragma unroll
  for (int r = 0; r < 8; ++r)
    C[(mbase + r) * Ncols + col] = acc[r];
}

// -------------------------------------------------------- scatter aggregate
// out[dst,:] += H[src,:] * rsqrt(deg[src]) * rsqrt(deg[dst]) (edges only;
// self-loop contribution is added in the finalize kernels as H[n,:]/deg[n]).
__global__ void agg_edges4_kernel(const int* __restrict__ edges, int E,
                                  const float* __restrict__ deg,
                                  const float* __restrict__ H,
                                  float* __restrict__ out, int F) {
  int gid = blockIdx.x * blockDim.x + threadIdx.x;
  int tpe = F >> 2;                 // threads per edge, 4 floats each
  int e = gid / tpe;
  if (e >= E) return;
  int fb = (gid - e * tpe) << 2;
  int s = edges[e];
  int d = edges[E + e];
  float nrm = rsqrtf(deg[s]) * rsqrtf(deg[d]);
  float4 hv = *(const float4*)(H + (size_t)s * F + fb);
  float* od = out + (size_t)d * F + fb;
  atomicAdd(&od[0], hv.x * nrm);
  atomicAdd(&od[1], hv.y * nrm);
  atomicAdd(&od[2], hv.z * nrm);
  atomicAdd(&od[3], hv.w * nrm);
}

// H_h = f16(relu(agg + XW/deg + b))
__global__ void finalize_relu_f16_kernel(const float* __restrict__ agg,
                                         const float* __restrict__ XW,
                                         const float* __restrict__ deg,
                                         const float* __restrict__ bias,
                                         _Float16* __restrict__ Hh, int F) {
  int i = blockIdx.x * blockDim.x + threadIdx.x;
  if (i >= N_NODES * F) return;
  int n = i / F, f = i - n * F;
  float v = agg[i] + XW[i] / deg[n] + bias[f];
  Hh[i] = (_Float16)fmaxf(v, 0.0f);
}

// emb = agg + HW/deg + b   (no relu on second conv)
__global__ void finalize_emb_kernel(const float* __restrict__ agg,
                                    const float* __restrict__ HW,
                                    const float* __restrict__ deg,
                                    const float* __restrict__ bias,
                                    float* __restrict__ emb, int F) {
  int i = blockIdx.x * blockDim.x + threadIdx.x;
  if (i >= N_NODES * F) return;
  int n = i / F, f = i - n * F;
  emb[i] = agg[i] + HW[i] / deg[n] + bias[f];
}

// ----------------------------------------------------------------- CNN head
// One block (64 threads, 2 waves) per scored edge.
// img [2,14,16] -> conv1(32,2x2)+pool -> [32,6,7] -> conv2(64,3x3)+pool+gmax
// -> [64] -> lin1+relu -> [32] -> lin2+sigmoid -> scalar.
__global__ void cnn_edge_kernel(const float* __restrict__ emb,     // [8,N,16]
                                const int* __restrict__ sl_pos,    // [2,SL_E]
                                const int* __restrict__ sl_neg,    // [2,SL_E]
                                const float* __restrict__ c1w,     // [32,2,2,2]
                                const float* __restrict__ c1b,     // [32]
                                const float* __restrict__ c2w,     // [64,32,3,3]
                                const float* __restrict__ c2b,     // [64]
                                const float* __restrict__ l1w,     // [32,64]
                                const float* __restrict__ l1b,     // [32]
                                const float* __restrict__ l2w,     // [1,32]
                                const float* __restrict__ l2b,     // [1]
                                float* __restrict__ out) {
  const int e = blockIdx.x;
  const int t = threadIdx.x;  // 64 threads

  __shared__ float img[2][14][16];
  __shared__ float cw1[256];
  __shared__ float cb1[32];
  __shared__ float pooled[32][6][7];
  __shared__ float feat[64];
  __shared__ float hidden[32];
  __shared__ int nodes[2];

  if (t < 2)
    nodes[t] = (e < SL_EDGES) ? sl_pos[t * SL_EDGES + e]
                              : sl_neg[t * SL_EDGES + (e - SL_EDGES)];
  for (int i = t; i < 256; i += 64) cw1[i] = c1w[i];
  if (t < 32) cb1[t] = c1b[t];
  __syncthreads();

  // emb_all stack order -> graph-bank index {sl=0, ppi=1, reactome=2, corum=3,
  // go_f=4, go_c=5, go_p=6, kegg=7}
  const int kmap[14] = {0, 2, 0, 7, 0, 4, 0, 3, 0, 5, 0, 6, 0, 1};
  for (int i = t; i < 2 * 14 * 16; i += 64) {
    int c = i / 224, rem = i - c * 224, k = rem / 16, d = rem - k * 16;
    img[c][k][d] = emb[((size_t)kmap[k] * N_NODES + nodes[c]) * 16 + d];
  }
  __syncthreads();

  // conv1 (valid, 13x15) fused with 2x2 maxpool -> [32,6,7]
  for (int p = t; p < 32 * 6 * 7; p += 64) {
    int oc = p / 42, rem = p - oc * 42, ph = rem / 7, pw = rem - ph * 7;
    float m = -INFINITY;
    for (int dh = 0; dh < 2; ++dh)
      for (int dw = 0; dw < 2; ++dw) {
        int h = 2 * ph + dh, w = 2 * pw + dw;
        float s = cb1[oc];
#pragma unroll
        for (int ic = 0; ic < 2; ++ic)
#pragma unroll
          for (int kh = 0; kh < 2; ++kh)
#pragma unroll
            for (int kw = 0; kw < 2; ++kw)
              s += img[ic][h + kh][w + kw] * cw1[oc * 8 + ic * 4 + kh * 2 + kw];
        m = fmaxf(m, s);
      }
    pooled[oc][ph][pw] = m;
  }
  __syncthreads();

  // conv2 + pool2 + global max: only positions (h 0..3, w 0..3) survive.
  {
    const int oc = t;
    float acc[16];
    float b = c2b[oc];
#pragma unroll
    for (int i = 0; i < 16; ++i) acc[i] = b;
    for (int ic = 0; ic < 32; ++ic)
#pragma unroll
      for (int kh = 0; kh < 3; ++kh)
#pragma unroll
        for (int kw = 0; kw < 3; ++kw) {
          float wt = c2w[oc * 288 + ic * 9 + kh * 3 + kw];
#pragma unroll
          for (int h = 0; h < 4; ++h)
#pragma unroll
            for (int w = 0; w < 4; ++w)
              acc[h * 4 + w] += pooled[ic][h + kh][w + kw] * wt;
        }
    float m = acc[0];
#pragma unroll
    for (int i = 1; i < 16; ++i) m = fmaxf(m, acc[i]);
    feat[oc] = m;
  }
  __syncthreads();

  if (t < 32) {
    float s = l1b[t];
    for (int i = 0; i < 64; ++i) s += feat[i] * l1w[t * 64 + i];
    hidden[t] = fmaxf(s, 0.0f);
  }
  __syncthreads();

  if (t == 0) {
    float s = l2b[0];
    for (int i = 0; i < 32; ++i) s += hidden[i] * l2w[i];
    out[e] = 1.0f / (1.0f + expf(-s));
  }
}

// ------------------------------------------------------------------- driver

static inline char* ws_take(char*& p, size_t bytes) {
  uintptr_t v = (uintptr_t)p;
  v = (v + 255) & ~(uintptr_t)255;
  char* r = (char*)v;
  p = r + bytes;
  return r;
}

extern "C" void kernel_launch(void* const* d_in, const int* in_sizes, int n_in,
                              void* d_out, int out_size, void* d_ws, size_t ws_size,
                              hipStream_t stream) {
  (void)in_sizes; (void)n_in; (void)out_size; (void)ws_size;
  const float* x       = (const float*)d_in[0];
  const int*   sl_pos  = (const int*)d_in[1];
  const int*   sl_neg  = (const int*)d_in[2];
  const int*   kg      = (const int*)d_in[3];
  const float* Wsl1    = (const float*)d_in[4];
  const float* bsl1    = (const float*)d_in[5];
  const float* Wsl2    = (const float*)d_in[6];
  const float* bsl2    = (const float*)d_in[7];
  const float* Wkg1    = (const float*)d_in[8];
  const float* bkg1    = (const float*)d_in[9];
  const float* Wkg2    = (const float*)d_in[10];
  const float* bkg2    = (const float*)d_in[11];
  const float* c1w     = (const float*)d_in[12];
  const float* c1b     = (const float*)d_in[13];
  const float* c2w     = (const float*)d_in[14];
  const float* c2b     = (const float*)d_in[15];
  const float* l1w     = (const float*)d_in[16];
  const float* l1b     = (const float*)d_in[17];
  const float* l2w     = (const float*)d_in[18];
  const float* l2b     = (const float*)d_in[19];
  float* y             = (float*)d_out;

  char* p = (char*)d_ws;
  _Float16* x_h  = (_Float16*)ws_take(p, (size_t)N_NODES * D_IN * 2);
  _Float16* w1f  = (_Float16*)ws_take(p, (size_t)8 * D_IN * 128 * 2);  // packed frags
  _Float16* w2f  = (_Float16*)ws_take(p, (size_t)8 * 128 * 16 * 2);    // packed frags
  float*    XW   = (float*)ws_take(p, (size_t)N_NODES * 128 * 4);
  float*    agg1 = (float*)ws_take(p, (size_t)N_NODES * 128 * 4);
  _Float16* H_h  = (_Float16*)ws_take(p, (size_t)N_NODES * 128 * 2);
  float*    HW   = (float*)ws_take(p, (size_t)N_NODES * 16 * 4);
  float*    agg2 = (float*)ws_take(p, (size_t)N_NODES * 16 * 4);
  float*    deg  = (float*)ws_take(p, (size_t)N_NODES * 4);
  float*    emb  = (float*)ws_take(p, (size_t)8 * N_NODES * 16 * 4);

  const int TB = 256;
  auto blocks = [](long n, int tb) { return dim3((unsigned)((n + tb - 1) / tb)); };

  // --- x -> f16 and weights -> packed WMMA fragments (once per launch) ---
  cvt_f32_to_f16_kernel<<<blocks((long)N_NODES * D_IN, TB), TB, 0, stream>>>(
      x, x_h, N_NODES * D_IN);
  for (int g = 0; g < 8; ++g) {
    const float* W1 = (g == 0) ? Wsl1 : (Wkg1 + (size_t)(g - 1) * D_IN * 128);
    const float* W2 = (g == 0) ? Wsl2 : (Wkg2 + (size_t)(g - 1) * 128 * 16);
    pack_b_frags_kernel<<<blocks(D_IN * 128, TB), TB, 0, stream>>>(
        W1, w1f + (size_t)g * D_IN * 128, D_IN, 128);
    pack_b_frags_kernel<<<blocks(128 * 16, TB), TB, 0, stream>>>(
        W2, w2f + (size_t)g * 128 * 16, 128, 16);
  }

  // --- 8 GCN encoders: g=0 is SL graph, g=1..7 are KG graphs 0..6 ---
  for (int g = 0; g < 8; ++g) {
    const int* edges = (g == 0) ? sl_pos : (kg + (size_t)(g - 1) * 2 * KG_EDGES);
    const int  E     = (g == 0) ? SL_EDGES : KG_EDGES;
    const float* b1  = (g == 0) ? bsl1 : (bkg1 + (g - 1) * 128);
    const float* b2  = (g == 0) ? bsl2 : (bkg2 + (g - 1) * 16);
    const _Float16* W1f = w1f + (size_t)g * D_IN * 128;
    const _Float16* W2f = w2f + (size_t)g * 128 * 16;

    // degrees (self loop contributes the initial 1.0)
    fill_f32_kernel<<<blocks(N_NODES, TB), TB, 0, stream>>>(deg, 1.0f, N_NODES);
    deg_count_kernel<<<blocks(E, TB), TB, 0, stream>>>(edges, E, deg);

    // layer 1: XW = x @ W1 (WMMA), aggregate, +self-loop +bias, relu, ->f16
    gemm_f16_wmma_kernel<<<dim3(N_NODES / 16, 128 / 16), 32, 0, stream>>>(
        x_h, W1f, XW, D_IN, 128);
    fill_f32_kernel<<<blocks((long)N_NODES * 128, TB), TB, 0, stream>>>(
        agg1, 0.0f, N_NODES * 128);
    agg_edges4_kernel<<<blocks((long)E * 32, TB), TB, 0, stream>>>(
        edges, E, deg, XW, agg1, 128);
    finalize_relu_f16_kernel<<<blocks((long)N_NODES * 128, TB), TB, 0, stream>>>(
        agg1, XW, deg, b1, H_h, 128);

    // layer 2: HW = H @ W2 (WMMA), aggregate, +self-loop +bias -> emb bank
    gemm_f16_wmma_kernel<<<dim3(N_NODES / 16, 1), 32, 0, stream>>>(
        H_h, W2f, HW, 128, 16);
    fill_f32_kernel<<<blocks((long)N_NODES * 16, TB), TB, 0, stream>>>(
        agg2, 0.0f, N_NODES * 16);
    agg_edges4_kernel<<<blocks((long)E * 4, TB), TB, 0, stream>>>(
        edges, E, deg, HW, agg2, 16);
    finalize_emb_kernel<<<blocks((long)N_NODES * 16, TB), TB, 0, stream>>>(
        agg2, HW, deg, b2, emb + (size_t)g * N_NODES * 16, 16);
  }

  // --- CNN head over all 40000 scored edges ---
  cnn_edge_kernel<<<dim3(NE_OUT), 64, 0, stream>>>(
      emb, sl_pos, sl_neg, c1w, c1b, c2w, c2b, l1w, l1b, l2w, l2b, y);
}
